// SparseAttentionBlock_70196945486469
// MI455X (gfx1250) — matrix operand order repaired and checked by reading, not compiled, not run on hardware
//
#include <hip/hip_runtime.h>

#define D_MODEL 1024
#define NUM_HEADS 16
#define HEAD_DIM 64
#define WINDOW 64
#define GLOBAL_TOKENS 4
#define RANDOM_TOKENS 32
#define BATCH 2
#define SEQ 2048
#define LN_EPS 1e-5f

typedef __bf16 bf16;
typedef __attribute__((ext_vector_type(16))) __bf16 v16bf;
typedef __attribute__((ext_vector_type(8)))  float  v8f;
typedef int v4i __attribute__((vector_size(16)));

// ---------------- CDNA5 async global->LDS path (guarded, fallback = sync copy)
#if defined(__has_builtin)
#  if __has_builtin(__builtin_amdgcn_global_load_async_to_lds_b128)
#    define HAS_ASYNC 1
#  endif
#endif
#ifndef HAS_ASYNC
#  define HAS_ASYNC 0
#endif

#if HAS_ASYNC
typedef __attribute__((address_space(1))) v4i* gv4i_p;
typedef __attribute__((address_space(3))) v4i* lv4i_p;
#endif

__device__ inline void copy16(const bf16* g, bf16* l) {
#if HAS_ASYNC
    __builtin_amdgcn_global_load_async_to_lds_b128((gv4i_p)g, (lv4i_p)l, 0, 0);
#else
    *(uint4*)l = *(const uint4*)g;
#endif
}

__device__ inline void wait_async0() {
#if HAS_ASYNC
#  if __has_builtin(__builtin_amdgcn_s_wait_asynccnt)
    __builtin_amdgcn_s_wait_asynccnt(0);
#  else
    asm volatile("s_wait_asynccnt 0x0" ::: "memory");
#  endif
#endif
}
__device__ inline void wait_async4() {
#if HAS_ASYNC
#  if __has_builtin(__builtin_amdgcn_s_wait_asynccnt)
    __builtin_amdgcn_s_wait_asynccnt(4);
#  else
    asm volatile("s_wait_asynccnt 0x4" ::: "memory");
#  endif
#endif
}

__device__ inline v8f wmma_bf16(v16bf a, v16bf b, v8f c) {
    return __builtin_amdgcn_wmma_f32_16x16x32_bf16(
        false, a, false, b, (short)0, c, false, false);
}

__device__ inline v16bf load16(const bf16* p) {
    union { v16bf v; uint4 q[2]; } u;
    u.q[0] = *(const uint4*)(p);
    u.q[1] = *(const uint4*)(p + 8);
    return u.v;
}

// ---------------------------------------------------------------- LayerNorm
__global__ __launch_bounds__(256)
void ln_kernel(const float* __restrict__ x, const float* __restrict__ gamma,
               const float* __restrict__ beta, bf16* __restrict__ xn) {
    const int row = blockIdx.x;
    const float* xr = x + (size_t)row * D_MODEL;
    __shared__ float red[256];

    float vals[4];
    float s = 0.f;
#pragma unroll
    for (int i = 0; i < 4; ++i) { vals[i] = xr[threadIdx.x + 256 * i]; s += vals[i]; }
    red[threadIdx.x] = s; __syncthreads();
    for (int off = 128; off > 0; off >>= 1) {
        if (threadIdx.x < off) red[threadIdx.x] += red[threadIdx.x + off];
        __syncthreads();
    }
    const float mu = red[0] * (1.0f / D_MODEL);
    __syncthreads();

    float ss = 0.f;
#pragma unroll
    for (int i = 0; i < 4; ++i) { float d = vals[i] - mu; ss += d * d; }
    red[threadIdx.x] = ss; __syncthreads();
    for (int off = 128; off > 0; off >>= 1) {
        if (threadIdx.x < off) red[threadIdx.x] += red[threadIdx.x + off];
        __syncthreads();
    }
    const float rstd = rsqrtf(red[0] * (1.0f / D_MODEL) + LN_EPS);

#pragma unroll
    for (int i = 0; i < 4; ++i) {
        int c = threadIdx.x + 256 * i;
        float y = (vals[i] - mu) * rstd * gamma[c] + beta[c];
        xn[(size_t)row * D_MODEL + c] = (bf16)y;
    }
}

// ------------------------------------------- f32 KxN -> bf16 NxK (transpose)
__global__ __launch_bounds__(256)
void transpose_f2bf_kernel(const float* __restrict__ src, bf16* __restrict__ dst,
                           int K, int N) {
    __shared__ bf16 tile[32][33];
    const int n0 = blockIdx.x * 32, k0 = blockIdx.y * 32;
    const int tx = threadIdx.x & 31, ty = threadIdx.x >> 5;   // 32 x 8
#pragma unroll
    for (int i = 0; i < 4; ++i)
        tile[ty + 8 * i][tx] = (bf16)src[(size_t)(k0 + ty + 8 * i) * N + n0 + tx];
    __syncthreads();
#pragma unroll
    for (int i = 0; i < 4; ++i)
        dst[(size_t)(n0 + ty + 8 * i) * K + k0 + tx] = tile[tx][ty + 8 * i];
}

// ---------------------------------------------------------------- random mask bitmap
__global__ __launch_bounds__(256)
void zero_kernel(unsigned* __restrict__ p, int n) {
    int i = blockIdx.x * 256 + threadIdx.x;
    if (i < n) p[i] = 0u;
}

__global__ __launch_bounds__(RANDOM_TOKENS)
void rnd_scatter_kernel(const int* __restrict__ rand_idx, unsigned* __restrict__ bits) {
    int i = blockIdx.x;
    int r = threadIdx.x;
    int j = rand_idx[i * RANDOM_TOKENS + r];
    atomicOr(&bits[(size_t)i * (SEQ / 32) + (j >> 5)], 1u << (j & 31));
}

// ---------------------------------------------------------------- bf16 GEMM
// C(MxN) = A(MxK, row-major) @ BT(NxK, row-major)^T
// Block tile 128x128, BK=32, double-buffered async LDS staging.
// 8 waves in 4x2 grid, wave tile 32x64 (2x4 WMMA tiles).
// MODE 0: store bf16 C. MODE 1: store f32 C + residual.
// MODE 2: store bf16 C (qkv) and scatter V-third into vt[b][h][d][seq].
template <int MODE>
__global__ __launch_bounds__(256)
void gemm_kernel(const bf16* __restrict__ A, const bf16* __restrict__ BT,
                 int M, int N, int K,
                 const float* __restrict__ residual,
                 float* __restrict__ Cf, bf16* __restrict__ Cb,
                 bf16* __restrict__ vt) {
    __shared__ bf16 As[2][128][32];
    __shared__ bf16 Bs[2][128][32];

    const int tid  = threadIdx.x;
    const int wave = tid >> 5;
    const int lane = tid & 31;
    const int lh   = lane >> 4;
    const int ln   = lane & 15;

    const int bm = blockIdx.x;
    const int bn = blockIdx.y;
    const int m0 = (wave >> 1) * 32;
    const int n0 = (wave & 1) * 64;

    v8f acc[2][4];
#pragma unroll
    for (int i = 0; i < 2; ++i)
#pragma unroll
        for (int j = 0; j < 4; ++j) acc[i][j] = (v8f){0.f,0.f,0.f,0.f,0.f,0.f,0.f,0.f};

    auto stage = [&](int buf, int k0) {
#pragma unroll
        for (int t = 0; t < 2; ++t) {
            int e = tid * 2 + t;
            int row = e >> 2, q = e & 3;
            copy16(A  + (size_t)(bm * 128 + row) * K + k0 + q * 8, &As[buf][row][q * 8]);
            copy16(BT + (size_t)(bn * 128 + row) * K + k0 + q * 8, &Bs[buf][row][q * 8]);
        }
    };

    const int NK = K / 32;
    stage(0, 0);

    for (int it = 0; it < NK; ++it) {
        const int cur = it & 1;
        const bool hasnext = (it + 1) < NK;
        if (hasnext) stage(1 - cur, (it + 1) * 32);     // prefetch next tile (async)
        if (hasnext) wait_async4(); else wait_async0(); // wait current tile only
        __syncthreads();

        v16bf af[2], bfr[4];
#pragma unroll
        for (int rt = 0; rt < 2; ++rt) {
            const bf16* ap = &As[cur][m0 + rt * 16 + ln][0];
            union { v16bf v; uint4 q[2]; } u;
            u.q[0] = *(const uint4*)(ap + lh * 16);
            u.q[1] = *(const uint4*)(ap + lh * 16 + 8);
            af[rt] = u.v;
        }
#pragma unroll
        for (int ct = 0; ct < 4; ++ct)
            bfr[ct] = load16(&Bs[cur][n0 + ct * 16 + ln][0] + lh * 16);

#pragma unroll
        for (int rt = 0; rt < 2; ++rt)
#pragma unroll
            for (int ct = 0; ct < 4; ++ct)
                acc[rt][ct] = wmma_bf16(af[rt], bfr[ct], acc[rt][ct]);
        __syncthreads();
    }

#pragma unroll
    for (int rt = 0; rt < 2; ++rt)
#pragma unroll
        for (int ct = 0; ct < 4; ++ct)
#pragma unroll
            for (int r = 0; r < 8; ++r) {
                int row = bm * 128 + m0 + rt * 16 + r + 8 * lh;
                int col = bn * 128 + n0 + ct * 16 + ln;
                size_t idx = (size_t)row * N + col;
                float v = acc[rt][ct][r];
                if (MODE == 1) {
                    Cf[idx] = v + residual[idx];
                } else {
                    Cb[idx] = (bf16)v;
                    if (MODE == 2 && col >= 2 * D_MODEL) {
                        int d  = col - 2 * D_MODEL;
                        int h  = d >> 6, dd = d & 63;
                        int b  = row >> 11, l = row & (SEQ - 1);
                        vt[(((size_t)b * NUM_HEADS + h) * HEAD_DIM + dd) * SEQ + l] =
                            (bf16)v;
                    }
                }
            }
}

// ---------------------------------------------------------------- sparse flash attention
__global__ __launch_bounds__(128)
void attn_kernel(const bf16* __restrict__ qkv, const bf16* __restrict__ vt,
                 const unsigned* __restrict__ rbits, bf16* __restrict__ attn_out) {
    const int qt   = blockIdx.x;
    const int h    = blockIdx.y;
    const int b    = blockIdx.z;
    const int tid  = threadIdx.x;
    const int wave = tid >> 5;
    const int lane = tid & 31;
    const int lh   = lane >> 4;
    const int ln   = lane & 15;

    __shared__ bf16 k_s[64][64];          // [key][kd]
    __shared__ bf16 v_t[64][64];          // [d][key]
    __shared__ bf16 p_s[4][16][64];
    __shared__ unsigned rb_s[4][16][2];

    const int    qrow0   = qt * 64 + wave * 16;
    const size_t rowbase = (size_t)b * SEQ;
    const int    NQKV    = 3 * D_MODEL;
    const int    qcol    = h * HEAD_DIM;
    const int    kcol    = D_MODEL + h * HEAD_DIM;
    const bf16*  vhead   = vt + ((size_t)b * NUM_HEADS + h) * HEAD_DIM * SEQ;

    v16bf qa[2];
    {
        const bf16* qptr = qkv + (rowbase + qrow0 + ln) * NQKV + qcol;
#pragma unroll
        for (int ks = 0; ks < 2; ++ks) {
            v16bf f;
#pragma unroll
            for (int e = 0; e < 16; ++e) {
                int k = ks * 32 + lh * 16 + e;
                f[e] = (bf16)(0.375f * (float)qptr[k]);   // fold 3/sqrt(64)
            }
            qa[ks] = f;
        }
    }

    float m_i[8], l_i[8];
#pragma unroll
    for (int r = 0; r < 8; ++r) { m_i[r] = -3.0e38f; l_i[r] = 0.f; }
    v8f oacc[4];
#pragma unroll
    for (int ct = 0; ct < 4; ++ct) oacc[ct] = (v8f){0.f,0.f,0.f,0.f,0.f,0.f,0.f,0.f};

    for (int jt = 0; jt <= qt; ++jt) {
        // K tile: 64 rows x 64 halves, contiguous 16B segments -> async
        for (int t = tid; t < 512; t += 128) {
            int r = t >> 3, q = t & 7;
            copy16(qkv + (rowbase + jt * 64 + r) * NQKV + kcol + q * 8, &k_s[r][q * 8]);
        }
        // V tile from pre-transposed vt: rows are d, contiguous over seq -> async
        for (int t = tid; t < 512; t += 128) {
            int r = t >> 3, q = t & 7;
            copy16(vhead + (size_t)r * SEQ + jt * 64 + q * 8, &v_t[r][q * 8]);
        }
        {
            int r = lane >> 1, wsel = lane & 1;
            rb_s[wave][r][wsel] =
                rbits[(size_t)(qt * 64 + wave * 16 + r) * (SEQ / 32) + (jt * 2 + wsel)];
        }
        wait_async0();
        __syncthreads();

        // S = Q @ K^T
        v8f sc[4];
#pragma unroll
        for (int ct = 0; ct < 4; ++ct) {
            const bf16* kb = &k_s[ct * 16 + ln][0];
            v8f z = (v8f){0.f,0.f,0.f,0.f,0.f,0.f,0.f,0.f};
            z = wmma_bf16(qa[0], load16(kb + lh * 16), z);
            z = wmma_bf16(qa[1], load16(kb + 32 + lh * 16), z);
            sc[ct] = z;
        }

        // mask counts + online softmax
        float sv[4][8], tmax[8];
#pragma unroll
        for (int r = 0; r < 8; ++r) tmax[r] = -3.0e38f;
#pragma unroll
        for (int ct = 0; ct < 4; ++ct) {
            int j = jt * 64 + ct * 16 + ln;
#pragma unroll
            for (int r = 0; r < 8; ++r) {
                int i = qrow0 + r + 8 * lh;
                int m = 0;
                if (j <= i) {
                    if (i - j <= WINDOW) ++m;
                    if (j < GLOBAL_TOKENS || j >= SEQ - GLOBAL_TOKENS) ++m;
                    unsigned w = rb_s[wave][r + 8 * lh][ct >> 1];
                    if ((w >> ((ct & 1) * 16 + ln)) & 1u) ++m;
                }
                float s = sc[ct][r] + (float)(m - 3) * 1.0e9f;
                sv[ct][r] = s;
                tmax[r] = fmaxf(tmax[r], s);
            }
        }
#pragma unroll
        for (int r = 0; r < 8; ++r)
#pragma unroll
            for (int off = 1; off < 16; off <<= 1)
                tmax[r] = fmaxf(tmax[r], __shfl_xor(tmax[r], off, 16));

        float alpha[8], rsum[8];
#pragma unroll
        for (int r = 0; r < 8; ++r) {
            float nm = fmaxf(m_i[r], tmax[r]);
            alpha[r] = __expf(m_i[r] - nm);
            m_i[r]   = nm;
            rsum[r]  = 0.f;
        }
#pragma unroll
        for (int ct = 0; ct < 4; ++ct)
#pragma unroll
            for (int r = 0; r < 8; ++r) {
                float p = __expf(sv[ct][r] - m_i[r]);
                rsum[r] += p;
                p_s[wave][r + 8 * lh][ct * 16 + ln] = (bf16)p;
            }
#pragma unroll
        for (int r = 0; r < 8; ++r) {
#pragma unroll
            for (int off = 1; off < 16; off <<= 1)
                rsum[r] += __shfl_xor(rsum[r], off, 16);
            l_i[r] = l_i[r] * alpha[r] + rsum[r];
        }
#pragma unroll
        for (int ct = 0; ct < 4; ++ct)
#pragma unroll
            for (int r = 0; r < 8; ++r) oacc[ct][r] *= alpha[r];

        // O += P @ V
        {
            const bf16* prow = &p_s[wave][ln][0];
#pragma unroll
            for (int ks = 0; ks < 2; ++ks) {
                union { v16bf v; uint4 q[2]; } u;
                u.q[0] = *(const uint4*)(prow + ks * 32 + lh * 16);
                u.q[1] = *(const uint4*)(prow + ks * 32 + lh * 16 + 8);
                v16bf pa = u.v;
#pragma unroll
                for (int ct = 0; ct < 4; ++ct) {
                    v16bf vb = load16(&v_t[ct * 16 + ln][ks * 32 + lh * 16]);
                    oacc[ct] = wmma_bf16(pa, vb, oacc[ct]);
                }
            }
        }
        __syncthreads();
    }

#pragma unroll
    for (int r = 0; r < 8; ++r) {
        float inv = 1.0f / l_i[r];
        int row = qrow0 + r + 8 * lh;
#pragma unroll
        for (int ct = 0; ct < 4; ++ct) {
            int d = ct * 16 + ln;
            attn_out[(rowbase + row) * D_MODEL + h * HEAD_DIM + d] =
                (bf16)(oacc[ct][r] * inv);
        }
    }
}

// ---------------------------------------------------------------- launch
extern "C" void kernel_launch(void* const* d_in, const int* in_sizes, int n_in,
                              void* d_out, int out_size, void* d_ws, size_t ws_size,
                              hipStream_t stream) {
    const float* x        = (const float*)d_in[0];
    const float* w_qkv    = (const float*)d_in[1];
    const float* w_out    = (const float*)d_in[2];
    const float* ln_gamma = (const float*)d_in[3];
    const float* ln_beta  = (const float*)d_in[4];
    const int*   rand_idx = (const int*)d_in[5];
    float*       out      = (float*)d_out;

    char* ws = (char*)d_ws;
    bf16* xn       = (bf16*)ws; ws += (size_t)BATCH * SEQ * D_MODEL * 2;
    bf16* wqkv_T   = (bf16*)ws; ws += (size_t)3 * D_MODEL * D_MODEL * 2;   // [3D][D]
    bf16* wout_T   = (bf16*)ws; ws += (size_t)D_MODEL * D_MODEL * 2;       // [D][D]
    bf16* qkv      = (bf16*)ws; ws += (size_t)BATCH * SEQ * 3 * D_MODEL * 2;
    bf16* vt       = (bf16*)ws; ws += (size_t)BATCH * NUM_HEADS * HEAD_DIM * SEQ * 2;
    bf16* attn     = (bf16*)ws; ws += (size_t)BATCH * SEQ * D_MODEL * 2;
    unsigned* rbit = (unsigned*)ws;

    // 1) LayerNorm -> bf16
    ln_kernel<<<BATCH * SEQ, 256, 0, stream>>>(x, ln_gamma, ln_beta, xn);
    // 2) weights -> bf16, transposed to NxK for async row-major staging
    transpose_f2bf_kernel<<<dim3(3 * D_MODEL / 32, D_MODEL / 32), 256, 0, stream>>>(
        w_qkv, wqkv_T, D_MODEL, 3 * D_MODEL);
    transpose_f2bf_kernel<<<dim3(D_MODEL / 32, D_MODEL / 32), 256, 0, stream>>>(
        w_out, wout_T, D_MODEL, D_MODEL);
    // 3) random-mask bitmap
    zero_kernel<<<(SEQ * (SEQ / 32) + 255) / 256, 256, 0, stream>>>(rbit, SEQ * (SEQ / 32));
    rnd_scatter_kernel<<<SEQ, RANDOM_TOKENS, 0, stream>>>(rand_idx, rbit);
    // 4) QKV GEMM (4096 x 3072 x 1024), bf16 out + V scattered into vt
    gemm_kernel<2><<<dim3((BATCH * SEQ) / 128, (3 * D_MODEL) / 128), 256, 0, stream>>>(
        xn, wqkv_T, BATCH * SEQ, 3 * D_MODEL, D_MODEL, nullptr, nullptr, qkv, vt);
    // 5) sparse flash attention
    attn_kernel<<<dim3(SEQ / 64, NUM_HEADS, BATCH), 128, 0, stream>>>(qkv, vt, rbit, attn);
    // 6) output projection + residual (4096 x 1024 x 1024), f32 out
    gemm_kernel<1><<<dim3((BATCH * SEQ) / 128, D_MODEL / 128), 256, 0, stream>>>(
        attn, wout_T, BATCH * SEQ, D_MODEL, D_MODEL, x, out, nullptr, nullptr);
}
